// DPLRStateSpaceModel_54460185313714
// MI455X (gfx1250) — matrix operand order repaired and checked by reading, not compiled, not run on hardware
//
#include <hip/hip_runtime.h>

#define DDIM   1024
#define NSTATE 16
#define BSZ    4
#define LSEQ   2048
#define CHUNKS 16
#define CHLEN  128
#define MROWS  (BSZ*LSEQ)

typedef __attribute__((ext_vector_type(16))) __bf16 v16bf;
typedef __attribute__((ext_vector_type(8)))  __bf16 v8bf;
typedef __attribute__((ext_vector_type(8)))  float  v8f;

__device__ __forceinline__ v16bf frag16(const __bf16* p0, const __bf16* p1) {
  v8bf a = *(const v8bf*)p0;
  v8bf b = *(const v8bf*)p1;
  return __builtin_shufflevector(a, b, 0,1,2,3,4,5,6,7,8,9,10,11,12,13,14,15);
}

// C[M,N] = A[M,K] * W[N,K]^T  (torch Linear), fp32 in/out, split-bf16 WMMA inside.
__global__ void __launch_bounds__(256) gemm_bf16split(
    const float* __restrict__ A, const float* __restrict__ W,
    float* __restrict__ C, int M, int N, int K)
{
  __shared__ __align__(16) __bf16 sAhi[128*32];
  __shared__ __align__(16) __bf16 sAlo[128*32];
  __shared__ __align__(16) __bf16 sBhi[128*32];
  __shared__ __align__(16) __bf16 sBlo[128*32];

  const int tid  = threadIdx.x;
  const int lane = tid & 31;
  const int wave = tid >> 5;
  const int wm   = (wave & 3) * 32;   // wave tile: 32 rows (M)
  const int wn   = (wave >> 2) * 64;  //            64 cols (N)
  const int m0   = blockIdx.y * 128;
  const int n0   = blockIdx.x * 128;

  const v8f vzero = {0.f,0.f,0.f,0.f,0.f,0.f,0.f,0.f};
  v8f acc[2][4];
  #pragma unroll
  for (int mi = 0; mi < 2; ++mi)
    #pragma unroll
    for (int ni = 0; ni < 4; ++ni) acc[mi][ni] = vzero;

  for (int kc = 0; kc < K; kc += 32) {
    // stage 128x32 fp32 tiles of A and W into LDS as bf16 hi/lo
    #pragma unroll
    for (int s = 0; s < 4; ++s) {
      int slot = tid + s * 256;            // 0..1023 float4 slots
      int row  = slot >> 3;
      int k4   = (slot & 7) * 4;
      float4 va = *(const float4*)(A + (size_t)(m0 + row) * K + kc + k4);
      float4 vb = *(const float4*)(W + (size_t)(n0 + row) * K + kc + k4);
      int o = row * 32 + k4;
      float af[4] = {va.x, va.y, va.z, va.w};
      float bf[4] = {vb.x, vb.y, vb.z, vb.w};
      #pragma unroll
      for (int j = 0; j < 4; ++j) {
        __bf16 h = (__bf16)af[j];
        sAhi[o + j] = h;
        sAlo[o + j] = (__bf16)(af[j] - (float)h);
        __bf16 g = (__bf16)bf[j];
        sBhi[o + j] = g;
        sBlo[o + j] = (__bf16)(bf[j] - (float)g);
      }
    }
    __syncthreads();

    // prefetch next K-chunk (global_prefetch_b8)
    if (kc + 32 < K) {
      int prow = tid >> 1;
      int half = (tid & 1) * 16;
      __builtin_prefetch(A + (size_t)(m0 + prow) * K + kc + 32 + half, 0, 1);
      __builtin_prefetch(W + (size_t)(n0 + prow) * K + kc + 32 + half, 0, 1);
    }

    // --- build fragments per documented 16-bit layouts ---
    const int rsel = lane & 15;
    const int kbA  = (lane < 16) ? 0 : 8;   // A: elems 0..7 = K kb..kb+7, 8..15 = K kb+16..kb+23
    const int kbB  = (lane < 16) ? 0 : 8;   // B: elems 0..7 = K kb..kb+7, 8..15 = K kb+16..kb+23 (lane = column)

    v16bf ah[2], al[2];
    #pragma unroll
    for (int mi = 0; mi < 2; ++mi) {
      int r = wm + mi * 16 + rsel;
      ah[mi] = frag16(sAhi + r * 32 + kbA, sAhi + r * 32 + kbA + 16);
      al[mi] = frag16(sAlo + r * 32 + kbA, sAlo + r * 32 + kbA + 16);
    }
    v16bf bh[4], bl[4];
    #pragma unroll
    for (int ni = 0; ni < 4; ++ni) {
      int c = wn + ni * 16 + rsel;
      bh[ni] = frag16(sBhi + c * 32 + kbB, sBhi + c * 32 + kbB + 16);
      bl[ni] = frag16(sBlo + c * 32 + kbB, sBlo + c * 32 + kbB + 16);
    }

    #pragma unroll
    for (int mi = 0; mi < 2; ++mi)
      #pragma unroll
      for (int ni = 0; ni < 4; ++ni) {
        acc[mi][ni] = __builtin_amdgcn_wmma_f32_16x16x32_bf16(
            false, ah[mi], false, bh[ni], (short)0, acc[mi][ni], false, false);
        acc[mi][ni] = __builtin_amdgcn_wmma_f32_16x16x32_bf16(
            false, ah[mi], false, bl[ni], (short)0, acc[mi][ni], false, false);
        acc[mi][ni] = __builtin_amdgcn_wmma_f32_16x16x32_bf16(
            false, al[mi], false, bh[ni], (short)0, acc[mi][ni], false, false);
      }
    __syncthreads();
  }

  // C/D layout: VGPR r -> row r (lanes 0-15) / row r+8 (lanes 16-31), col = lane&15
  const int rowOff = (lane < 16) ? 0 : 8;
  const int col    = lane & 15;
  #pragma unroll
  for (int mi = 0; mi < 2; ++mi)
    #pragma unroll
    for (int ni = 0; ni < 4; ++ni) {
      size_t base = (size_t)(m0 + wm + mi * 16 + rowOff) * N + (n0 + wn + ni * 16 + col);
      #pragma unroll
      for (int r = 0; r < 8; ++r)
        C[base + (size_t)r * N] = acc[mi][ni][r];
    }
}

// Ā = exp(dt*A), CB = C * (Ā-1)/A * B, Āpow = Ā^CHLEN
__global__ void __launch_bounds__(256) ssm_coeffs(
    const float* __restrict__ A_log, const float* __restrict__ B_ssm,
    const float* __restrict__ C_ssm, const float* __restrict__ dt_log,
    float* __restrict__ Abar, float* __restrict__ CBv, float* __restrict__ Apow)
{
  int g = blockIdx.x * 256 + threadIdx.x;
  if (g >= DDIM * NSTATE) return;
  int d = g >> 4;
  float Ad = -__expf(A_log[g]);
  float dt = __expf(dt_log[d]);
  float ab = __expf(dt * Ad);
  float bb = (ab - 1.0f) / Ad * B_ssm[g];
  Abar[g] = ab;
  CBv[g]  = C_ssm[g] * bb;
  Apow[g] = __expf(dt * Ad * (float)CHLEN);
}

// pass 1: per-(b,d,chunk) local scan from zero state -> chunk-final states
__global__ void __launch_bounds__(256) ssm_chunk_local(
    const float* __restrict__ xin, const float* __restrict__ Abar,
    const float* __restrict__ CBv, float* __restrict__ states)
{
  int d = blockIdx.x * 256 + threadIdx.x;
  int c = blockIdx.y;
  int b = blockIdx.z;
  float a[NSTATE], k[NSTATE], s[NSTATE];
  #pragma unroll
  for (int n = 0; n < NSTATE; ++n) {
    a[n] = Abar[d * NSTATE + n];
    k[n] = CBv[d * NSTATE + n];
    s[n] = 0.f;
  }
  const float* xp = xin + ((size_t)b * LSEQ + (size_t)c * CHLEN) * DDIM + d;
  for (int t = 0; t < CHLEN; ++t) {
    float x = xp[(size_t)t * DDIM];
    #pragma unroll
    for (int n = 0; n < NSTATE; ++n) s[n] = fmaf(a[n], s[n], k[n] * x);
  }
  float* sp = states + (((size_t)b * CHUNKS + c) * DDIM + d) * NSTATE;
  #pragma unroll
  for (int n = 0; n < NSTATE; ++n) sp[n] = s[n];
}

// pass 2: serial prefix over the 16 chunks, in place -> chunk initial states
__global__ void __launch_bounds__(256) ssm_prefix(
    const float* __restrict__ Apow, float* __restrict__ states)
{
  int g = blockIdx.x * 256 + threadIdx.x;  // over B*D*N = 65536
  int n = g & (NSTATE - 1);
  int d = (g >> 4) & (DDIM - 1);
  int b = g >> 14;
  float ap = Apow[d * NSTATE + n];
  float p = 0.f;
  for (int c = 0; c < CHUNKS; ++c) {
    size_t idx = (((size_t)b * CHUNKS + c) * DDIM + d) * NSTATE + n;
    float local = states[idx];
    states[idx] = p;              // initial state for chunk c
    p = fmaf(ap, p, local);
  }
}

// pass 3: re-scan each chunk from its true initial state, emit y + D*x
__global__ void __launch_bounds__(256) ssm_chunk_final(
    const float* __restrict__ xin, const float* __restrict__ Abar,
    const float* __restrict__ CBv, const float* __restrict__ states,
    const float* __restrict__ Dssm, float* __restrict__ y)
{
  int d = blockIdx.x * 256 + threadIdx.x;
  int c = blockIdx.y;
  int b = blockIdx.z;
  float a[NSTATE], k[NSTATE], s[NSTATE];
  const float* sp = states + (((size_t)b * CHUNKS + c) * DDIM + d) * NSTATE;
  #pragma unroll
  for (int n = 0; n < NSTATE; ++n) {
    a[n] = Abar[d * NSTATE + n];
    k[n] = CBv[d * NSTATE + n];
    s[n] = sp[n];
  }
  float dsk = Dssm[d];
  const float* xp = xin + ((size_t)b * LSEQ + (size_t)c * CHLEN) * DDIM + d;
  float*       yp = y   + ((size_t)b * LSEQ + (size_t)c * CHLEN) * DDIM + d;
  for (int t = 0; t < CHLEN; ++t) {
    float x = xp[(size_t)t * DDIM];
    float accv = dsk * x;
    #pragma unroll
    for (int n = 0; n < NSTATE; ++n) {
      s[n] = fmaf(a[n], s[n], k[n] * x);
      accv += s[n];
    }
    yp[(size_t)t * DDIM] = accv;
  }
}

extern "C" void kernel_launch(void* const* d_in, const int* in_sizes, int n_in,
                              void* d_out, int out_size, void* d_ws, size_t ws_size,
                              hipStream_t stream)
{
  (void)in_sizes; (void)n_in; (void)out_size; (void)ws_size;
  const float* x      = (const float*)d_in[0];
  const float* W_in   = (const float*)d_in[1];
  const float* W_out  = (const float*)d_in[2];
  const float* A_log  = (const float*)d_in[3];
  const float* B_ssm  = (const float*)d_in[4];
  const float* C_ssm  = (const float*)d_in[5];
  const float* dt_log = (const float*)d_in[6];
  const float* D_ssm  = (const float*)d_in[7];
  float* out = (float*)d_out;

  float* ws   = (float*)d_ws;
  float* Abar = ws;
  float* CBv  = Abar + DDIM * NSTATE;
  float* Apow = CBv  + DDIM * NSTATE;
  float* xin  = Apow + DDIM * NSTATE;                       // 8192*1024 f32
  float* st   = xin  + (size_t)MROWS * DDIM;                // B*CHUNKS*D*N f32
  float* ybuf = st   + (size_t)BSZ * CHUNKS * DDIM * NSTATE;// 8192*1024 f32

  ssm_coeffs<<<(DDIM * NSTATE + 255) / 256, 256, 0, stream>>>(
      A_log, B_ssm, C_ssm, dt_log, Abar, CBv, Apow);

  dim3 ggrid(DDIM / 128, MROWS / 128);
  gemm_bf16split<<<ggrid, 256, 0, stream>>>(x, W_in, xin, MROWS, DDIM, DDIM);

  dim3 sgrid(DDIM / 256, CHUNKS, BSZ);
  ssm_chunk_local<<<sgrid, 256, 0, stream>>>(xin, Abar, CBv, st);
  ssm_prefix<<<(BSZ * DDIM * NSTATE) / 256, 256, 0, stream>>>(Apow, st);
  ssm_chunk_final<<<sgrid, 256, 0, stream>>>(xin, Abar, CBv, st, D_ssm, ybuf);

  gemm_bf16split<<<ggrid, 256, 0, stream>>>(ybuf, W_out, out, MROWS, DDIM, DDIM);
}
